// DotProductAttention_78219944395428
// MI455X (gfx1250) — matrix operand order repaired and checked by reading, not compiled, not run on hardware
//
#include <hip/hip_runtime.h>

typedef _Float16 f16;
typedef __attribute__((ext_vector_type(16))) _Float16 v16h;
typedef __attribute__((ext_vector_type(8)))  _Float16 v8h;
typedef __attribute__((ext_vector_type(4)))  _Float16 v4h;
typedef __attribute__((ext_vector_type(2)))  _Float16 v2h;
typedef __attribute__((ext_vector_type(8)))  float    v8f;

#define B_ 2
#define H_ 16
#define S_ 2048
#define D_ 64
#define VSTR 72    // padded Vt row stride (f16): 144B = 36 dw; rows stay 16B-aligned

union U16h { v16h v; f16 e[16]; v8h h[2]; v4h q[4]; };

// Packed xor-butterfly max over the 16 lanes of each half-wave: two rows'
// maxima travel in one dword (f16 pair). Softmax is shift-invariant, so f16
// precision of the max cancels exactly after normalization.
__device__ __forceinline__ v2h halfwave_rowmax_pk(v2h x) {
    union { v2h h; int i; } u, s;
    u.h = x; s.i = __builtin_amdgcn_ds_swizzle(u.i, 0x041f); // xor 1
    x = __builtin_elementwise_max(x, s.h);
    u.h = x; s.i = __builtin_amdgcn_ds_swizzle(u.i, 0x081f); // xor 2
    x = __builtin_elementwise_max(x, s.h);
    u.h = x; s.i = __builtin_amdgcn_ds_swizzle(u.i, 0x101f); // xor 4
    x = __builtin_elementwise_max(x, s.h);
    u.h = x; s.i = __builtin_amdgcn_ds_swizzle(u.i, 0x201f); // xor 8
    x = __builtin_elementwise_max(x, s.h);
    return x;
}

__global__ __launch_bounds__(128)
void fa_causal_f16wmma(const float* __restrict__ Q, const float* __restrict__ K,
                       const float* __restrict__ V, float* __restrict__ O) {
    __shared__ __attribute__((aligned(32))) f16 Kl[64 * 64];          // K chunk, row-major
    __shared__ __attribute__((aligned(32))) f16 Vt[64 * VSTR];        // V chunk, transposed
    __shared__ __attribute__((aligned(32))) f16 Pl[4][2][16 * 64];    // per-wave, per-subtile P

    const int nQ   = S_ / 128;
    const int bh   = blockIdx.x / nQ;
    const int qt   = blockIdx.x % nQ;
    const int q0   = qt * 128;
    const int tid  = threadIdx.x;
    const int w    = tid >> 5;             // 4 waves, each owns 32 q-rows
    const int lane = tid & 31;
    const int l16  = lane & 15;
    const int half = lane >> 4;
    const int qw   = q0 + w * 32;          // first q-row of this wave's strip

    const float* Qb = Q + (size_t)bh * S_ * D_;
    const float* Kb = K + (size_t)bh * S_ * D_;
    const float* Vb = V + (size_t)bh * S_ * D_;
    float*       Ob = O + (size_t)bh * S_ * D_;

    // (1/sqrt(D))*log2(e) folded into Q: QK scores emerge in the log2 domain.
    const float scale2 = 0.125f * 1.4426950408889634f;

    // Two 16-row A tiles per wave (u = 0,1), each split into 2 K=32 chunks.
    U16h qa[2][2];
    #pragma unroll
    for (int u = 0; u < 2; ++u) {
        const float* qrow = Qb + (size_t)(qw + u * 16 + l16) * D_;
        #pragma unroll
        for (int ch = 0; ch < 2; ++ch) {
            #pragma unroll
            for (int i = 0; i < 8; ++i) {
                qa[u][ch].e[i]     = (f16)(qrow[ch * 32 + half * 16 + i] * scale2);
                qa[u][ch].e[8 + i] = (f16)(qrow[ch * 32 + half * 16 + 8 + i] * scale2);
            }
        }
    }

    U16h ones;   // accl = P*ones: softmax denominator rides along with O
    #pragma unroll
    for (int i = 0; i < 16; ++i) ones.e[i] = (f16)1.0f;

    float m_i[2][8];
    v8f acc[2][4], accl[2];
    #pragma unroll
    for (int u = 0; u < 2; ++u) {
        #pragma unroll
        for (int r = 0; r < 8; ++r) { m_i[u][r] = -3.0e38f; accl[u][r] = 0.0f; }
        #pragma unroll
        for (int t = 0; t < 4; ++t) {
            #pragma unroll
            for (int r = 0; r < 8; ++r) acc[u][t][r] = 0.0f;
        }
    }

    // Single loop body (kept out-of-line once for I-cache): runtime mask flag,
    // taken only on the 1-2 diagonal chunks that actually straddle this wave.
    const int kend = q0 + 128;             // causal bound for this block's rows
    for (int kbase = 0; kbase < kend; kbase += 64) {
        __syncthreads();
        // ---- stage K row-major: float4 reads, packed b64 stores
        #pragma unroll
        for (int it = 0; it < 8; ++it) {
            const int j   = (it * 128 + tid) * 4;
            const int row = j >> 6, col = j & 63;
            const float4 k4 = *(const float4*)&Kb[(size_t)(kbase + row) * D_ + col];
            v4h kp; kp[0] = (f16)k4.x; kp[1] = (f16)k4.y; kp[2] = (f16)k4.z; kp[3] = (f16)k4.w;
            *(v4h*)&Kl[j] = kp;
        }
        // ---- stage V transposed in 4x4 blocks
        {
            const int c4 = (tid & 15) * 4;
            #pragma unroll
            for (int it = 0; it < 2; ++it) {
                const int r4 = (tid >> 4) * 4 + it * 32;
                const float4 a0 = *(const float4*)&Vb[(size_t)(kbase + r4 + 0) * D_ + c4];
                const float4 a1 = *(const float4*)&Vb[(size_t)(kbase + r4 + 1) * D_ + c4];
                const float4 a2 = *(const float4*)&Vb[(size_t)(kbase + r4 + 2) * D_ + c4];
                const float4 a3 = *(const float4*)&Vb[(size_t)(kbase + r4 + 3) * D_ + c4];
                v4h t0; t0[0] = (f16)a0.x; t0[1] = (f16)a1.x; t0[2] = (f16)a2.x; t0[3] = (f16)a3.x;
                v4h t1; t1[0] = (f16)a0.y; t1[1] = (f16)a1.y; t1[2] = (f16)a2.y; t1[3] = (f16)a3.y;
                v4h t2; t2[0] = (f16)a0.z; t2[1] = (f16)a1.z; t2[2] = (f16)a2.z; t2[3] = (f16)a3.z;
                v4h t3; t3[0] = (f16)a0.w; t3[1] = (f16)a1.w; t3[2] = (f16)a2.w; t3[3] = (f16)a3.w;
                *(v4h*)&Vt[(c4 + 0) * VSTR + r4] = t0;
                *(v4h*)&Vt[(c4 + 1) * VSTR + r4] = t1;
                *(v4h*)&Vt[(c4 + 2) * VSTR + r4] = t2;
                *(v4h*)&Vt[(c4 + 3) * VSTR + r4] = t3;
            }
        }
        __syncthreads();

        if (kbase > qw + 31) continue;     // strip above diagonal: staged for others only
        const bool masked = (kbase + 63) > qw;   // per-wave exact mask need

        // ---- scores: each K B-tile pair feeds both q-subtiles (4 WMMAs per load pair)
        v8f s[2][4];
        #pragma unroll
        for (int t = 0; t < 4; ++t) {
            const v16h b0 = *(const v16h*)&Kl[(t * 16 + l16) * 64 + half * 16];
            const v16h b1 = *(const v16h*)&Kl[(t * 16 + l16) * 64 + 32 + half * 16];
            #pragma unroll
            for (int u = 0; u < 2; ++u) {
                v8f z = {};
                z = __builtin_amdgcn_wmma_f32_16x16x32_f16(false, qa[u][0].v, false, b0, (short)0, z, false, false);
                z = __builtin_amdgcn_wmma_f32_16x16x32_f16(false, qa[u][1].v, false, b1, (short)0, z, false, false);
                s[u][t] = z;
            }
        }

        #pragma unroll
        for (int u = 0; u < 2; ++u) {
            if (masked) {
                #pragma unroll
                for (int r = 0; r < 8; ++r) {
                    const int rowi = qw + u * 16 + r + half * 8;
                    #pragma unroll
                    for (int t = 0; t < 4; ++t)
                        if (kbase + t * 16 + l16 > rowi) s[u][t][r] = -3.0e38f;
                }
            }
            // per-row max: local max3 tree, then packed cross-lane butterfly
            float mx[8];
            #pragma unroll
            for (int pr = 0; pr < 4; ++pr) {
                const int r0 = 2 * pr, r1 = 2 * pr + 1;
                const float a = fmaxf(fmaxf(s[u][0][r0], s[u][1][r0]), fmaxf(s[u][2][r0], s[u][3][r0]));
                const float b = fmaxf(fmaxf(s[u][0][r1], s[u][1][r1]), fmaxf(s[u][2][r1], s[u][3][r1]));
                v2h pk; pk[0] = (f16)a; pk[1] = (f16)b;
                pk = halfwave_rowmax_pk(pk);
                mx[r0] = (float)pk[0]; mx[r1] = (float)pk[1];
            }
            // online softmax update: vectorized alpha rescale + P spill
            v8f alphav;
            #pragma unroll
            for (int r = 0; r < 8; ++r) {
                const float mnew = fmaxf(m_i[u][r], mx[r]);
                alphav[r] = __builtin_exp2f(m_i[u][r] - mnew);
                m_i[u][r] = mnew;
            }
            accl[u] *= alphav;
            #pragma unroll
            for (int t = 0; t < 4; ++t) acc[u][t] *= alphav;
            #pragma unroll
            for (int r = 0; r < 8; ++r) {
                const int prow = r + half * 8;
                #pragma unroll
                for (int t = 0; t < 4; ++t)
                    Pl[w][u][prow * 64 + t * 16 + l16] = (f16)__builtin_exp2f(s[u][t][r] - m_i[u][r]);
            }
        }

        // ---- O += P*V, l += P*ones; each V B-tile pair feeds both q-subtiles
        U16h pa[2][2];
        #pragma unroll
        for (int u = 0; u < 2; ++u) {
            pa[u][0].h[0] = *(const v8h*)&Pl[w][u][l16 * 64 + half * 8];
            pa[u][0].h[1] = *(const v8h*)&Pl[w][u][l16 * 64 + 16 + half * 8];
            pa[u][1].h[0] = *(const v8h*)&Pl[w][u][l16 * 64 + 32 + half * 8];
            pa[u][1].h[1] = *(const v8h*)&Pl[w][u][l16 * 64 + 48 + half * 8];
            accl[u] = __builtin_amdgcn_wmma_f32_16x16x32_f16(false, pa[u][0].v, false, ones.v, (short)0, accl[u], false, false);
            accl[u] = __builtin_amdgcn_wmma_f32_16x16x32_f16(false, pa[u][1].v, false, ones.v, (short)0, accl[u], false, false);
        }
        #pragma unroll
        for (int dt = 0; dt < 4; ++dt) {
            U16h vb0, vb1;   // 16B-aligned rows (VSTR=72): b128 pairs
            vb0.h[0] = *(const v8h*)&Vt[(dt * 16 + l16) * VSTR + half * 8];
            vb0.h[1] = *(const v8h*)&Vt[(dt * 16 + l16) * VSTR + 16 + half * 8];
            vb1.h[0] = *(const v8h*)&Vt[(dt * 16 + l16) * VSTR + 32 + half * 8];
            vb1.h[1] = *(const v8h*)&Vt[(dt * 16 + l16) * VSTR + 48 + half * 8];
            #pragma unroll
            for (int u = 0; u < 2; ++u) {
                acc[u][dt] = __builtin_amdgcn_wmma_f32_16x16x32_f16(false, pa[u][0].v, false, vb0.v, (short)0, acc[u][dt], false, false);
                acc[u][dt] = __builtin_amdgcn_wmma_f32_16x16x32_f16(false, pa[u][1].v, false, vb1.v, (short)0, acc[u][dt], false, false);
            }
        }
    }

    // ---- epilogue: O /= l
    #pragma unroll
    for (int u = 0; u < 2; ++u) {
        #pragma unroll
        for (int r = 0; r < 8; ++r) {
            const float inv = 1.0f / accl[u][r];
            const int row = qw + u * 16 + r + half * 8;
            #pragma unroll
            for (int dt = 0; dt < 4; ++dt)
                Ob[(size_t)row * D_ + dt * 16 + l16] = acc[u][dt][r] * inv;
        }
    }
}

extern "C" void kernel_launch(void* const* d_in, const int* in_sizes, int n_in,
                              void* d_out, int out_size, void* d_ws, size_t ws_size,
                              hipStream_t stream) {
    (void)in_sizes; (void)n_in; (void)d_ws; (void)ws_size; (void)out_size;
    const float* Q = (const float*)d_in[0];
    const float* K = (const float*)d_in[1];
    const float* V = (const float*)d_in[2];
    // d_in[3] is the causal mask; causality is computed analytically.
    float* O = (float*)d_out;

    const int blocks = B_ * H_ * (S_ / 128);   // 512 workgroups of 4 waves x 32 rows
    fa_causal_f16wmma<<<blocks, 128, 0, stream>>>(Q, K, V, O);
}
